// LocalGridAlignAttention_52810917871721
// MI455X (gfx1250) — matrix operand order repaired and checked by reading, not compiled, not run on hardware
//
#include <hip/hip_runtime.h>
#include <hip/hip_bf16.h>

typedef __attribute__((ext_vector_type(16))) __bf16 v16bf;
typedef __attribute__((ext_vector_type(8)))  float  v8f;

#define SCALING 0.17677669529663687f   // 32^-0.5

// workspace layout (bytes)
#define WKP_OFF 0u
#define WVP_OFF 131072u
#define WQP_OFF 262144u
#define Q_OFF   393216u          // 8*256*1024 fp32 = 8388608 B
#define GW_OFF  8781824u         // 8*1024 fp32 = 32768 B

// ---------------------------------------------------------------------------
// A-fragment (16x32 bf16) lane/element mapping used in this file (contiguous k):
//   lane l: m = l&15 ; hi = l>>4
//   elem i: k = 16*hi + i
// B-fragment (32x16 bf16): lane l: n = l&15 ; hi = l>>4 ; elem i: k = 16*hi + i
// D (16x16 f32, v8f): lane l reg r: n = l&15 ; m = r + 8*(l>>4)
// ---------------------------------------------------------------------------

__global__ void __launch_bounds__(256) pack_weights_kernel(
    const float* __restrict__ wq, const float* __restrict__ wk,
    const float* __restrict__ wv,
    __bf16* __restrict__ wkp, __bf16* __restrict__ wvp, __bf16* __restrict__ wqp) {
  int gid = blockIdx.x * 256 + threadIdx.x;       // 3 * 65536 total
  int wid = gid >> 16;
  int e   = gid & 65535;
  int f = e >> 9;           // fragment 0..127  (f = mtile*8 + ktile)
  int l = (e >> 4) & 31;    // lane
  int i = e & 15;           // element within lane
  int m  = l & 15;
  int hi = l >> 4;
  int k  = 16 * hi + i;
  int o  = (f >> 3) * 16 + m;
  int c  = (f & 7) * 32 + k;
  const float* w = (wid == 0) ? wk : (wid == 1) ? wv : wq;
  float val = w[o * 256 + c];
  if (wid == 2) val *= SCALING;                   // fold q scaling into weights
  __bf16* dst = (wid == 0) ? wkp : (wid == 1) ? wvp : wqp;
  dst[e] = (__bf16)val;
}

// ---------------------------------------------------------------------------
// q = (s*Wq) x + s*bq   and  grid_weight = sigmoid(wg.x + bg)
// one WG = (batch b, 128-pixel block); x tile staged in LDS as B-fragments.
// ---------------------------------------------------------------------------
__global__ void __launch_bounds__(256) proj_q_gw_kernel(
    const float* __restrict__ x, const __bf16* __restrict__ wqp,
    const float* __restrict__ bq, const float* __restrict__ wg,
    const float* __restrict__ bg,
    float* __restrict__ qout, float* __restrict__ gwout) {
  extern __shared__ char dynsmem[];
  __bf16* Xl = (__bf16*)dynsmem;                  // 64 frags * 512 el = 64 KB
  int blk  = blockIdx.x;                          // 64 blocks
  int b    = blk >> 3;
  int pix0 = (blk & 7) * 128;
  int tid  = threadIdx.x;

  // stage x[b, :, pix0:pix0+128] -> LDS B-fragments (bf16)
  for (int it = 0; it < 32; ++it) {
    int c  = it * 8 + (tid >> 5);
    int p4 = (tid & 31) * 4;
    const float* src = &x[(size_t)(b * 256 + c) * 1024 + pix0 + p4];
    if (it + 2 < 32)
      __builtin_prefetch(src + 2 * 8 * 1024, 0, 1);   // next-next channel group
    const float4 v = *(const float4*)src;
    float vv[4] = {v.x, v.y, v.z, v.w};
    int fk = c >> 5, kk = c & 31;
    int lbase = (kk & 16) ? 16 : 0;
    int i = kk & 15;
#pragma unroll
    for (int u = 0; u < 4; ++u) {
      int p = p4 + u;
      Xl[(fk * 8 + (p >> 4)) * 512 + (lbase + (p & 15)) * 16 + i] = (__bf16)vv[u];
    }
  }
  __syncthreads();

  int wvid = tid >> 5, lane = tid & 31, hi = lane >> 4, nlo = lane & 15;

  // hoist A fragments (2 mtiles x 8 kfrags) and scaled biases into registers
  v16bf aq[2][8];
#pragma unroll
  for (int mm = 0; mm < 2; ++mm)
#pragma unroll
    for (int fk = 0; fk < 8; ++fk)
      aq[mm][fk] =
          *(const v16bf*)&wqp[(size_t)(((2 * wvid + mm) * 8 + fk) * 32 + lane) * 16];
  float bq0[8], bq1[8];
#pragma unroll
  for (int r = 0; r < 8; ++r) {
    bq0[r] = bq[(2 * wvid) * 16 + hi * 8 + r] * SCALING;
    bq1[r] = bq[(2 * wvid + 1) * 16 + hi * 8 + r] * SCALING;
  }

  // each wave: 2 mtiles x 8 ntiles, K = 256 (8 fragments)
  for (int nt = 0; nt < 8; ++nt) {
    v8f acc0 = {}; v8f acc1 = {};
#pragma unroll
    for (int fk = 0; fk < 8; ++fk) {
      v16bf bfrag = *(const v16bf*)&Xl[(fk * 8 + nt) * 512 + lane * 16];
      acc0 = __builtin_amdgcn_wmma_f32_16x16x32_bf16(false, aq[0][fk], false, bfrag,
                                                     (short)0, acc0, false, false);
      acc1 = __builtin_amdgcn_wmma_f32_16x16x32_bf16(false, aq[1][fk], false, bfrag,
                                                     (short)0, acc1, false, false);
    }
    int pix = pix0 + nt * 16 + nlo;
#pragma unroll
    for (int r = 0; r < 8; ++r) {
      int ch0 = (2 * wvid) * 16 + hi * 8 + r;
      int ch1 = (2 * wvid + 1) * 16 + hi * 8 + r;
      qout[(size_t)(b * 256 + ch0) * 1024 + pix] = acc0[r] + bq0[r];
      qout[(size_t)(b * 256 + ch1) * 1024 + pix] = acc1[r] + bq1[r];
    }
  }

  // grid weight gate (one thread per pixel, x read back from LDS fragments)
  if (tid < 128) {
    int p = tid;
    int fn = p >> 4, n = p & 15;
    float acc = bg[0];
    for (int c = 0; c < 256; ++c) {
      int fk = c >> 5, kk = c & 31;
      int l = ((kk & 16) ? 16 : 0) + n;
      acc += (float)Xl[(fk * 8 + fn) * 512 + l * 16 + (kk & 15)] * wg[c];
    }
    gwout[b * 1024 + pix0 + p] = 1.0f / (1.0f + __expf(-acc));
  }
}

// ---------------------------------------------------------------------------
// Fully fused local-grid attention. One WG = (batch b, 4x4 query block).
// featmap patch (256ch x 16x16 px) -> LDS once; K and V projections computed
// in-place with WMMA; featmap never re-read, k/v never hit HBM.
// LDS: F 128KB + KV 128KB + q 16KB + attn 8KB + gw 64B = 286784 B  (<320KB)
// ---------------------------------------------------------------------------
__global__ void __launch_bounds__(256) fused_attn_kernel(
    const float* __restrict__ featmap,
    const __bf16* __restrict__ wkp, const __bf16* __restrict__ wvp,
    const float* __restrict__ bk, const float* __restrict__ bv,
    const float* __restrict__ qbuf, const float* __restrict__ gwbuf,
    float* __restrict__ out) {
  extern __shared__ char dynsmem[];
  __bf16* Fl = (__bf16*)dynsmem;                         // 131072 B
  __bf16* KV = (__bf16*)(dynsmem + 131072);              // 131072 B
  float*  Ql = (float*)(dynsmem + 262144);               // 16384 B
  float*  At = (float*)(dynsmem + 262144 + 16384);       // 8192 B
  float*  Gl = (float*)(dynsmem + 262144 + 16384 + 8192);// 64 B

  int blk = blockIdx.x;                                  // 512 blocks
  int b = blk >> 6;
  int blkid = blk & 63;
  int hq0 = (blkid >> 3) * 4;
  int wq0 = (blkid & 7) * 4;
  int y0 = hq0 * 4, x0 = wq0 * 4;
  int tid = threadIdx.x;
  int wvid = tid >> 5, lane = tid & 31, hi = lane >> 4, nlo = lane & 15;

  // ---- stage featmap 16x16 patch (all 256 ch) as B-fragments, bf16 ----
  for (int it = 0; it < 64; ++it) {
    int c  = it * 4 + (tid >> 6);
    int p4 = (tid & 63) * 4;
    int fy = p4 >> 4, fx = p4 & 15;
    const float* src =
        &featmap[(size_t)((b * 256 + c) * 128 + y0 + fy) * 128 + x0 + fx];
    if (it + 2 < 64)
      __builtin_prefetch(src + (size_t)2 * 4 * 128 * 128, 0, 1);
    const float4 v = *(const float4*)src;
    float vv[4] = {v.x, v.y, v.z, v.w};
    int fk = c >> 5, kk = c & 31;
    int lbase = (kk & 16) ? 16 : 0;
    int i = kk & 15;
#pragma unroll
    for (int u = 0; u < 4; ++u) {
      int p = p4 + u;
      Fl[(fk * 16 + (p >> 4)) * 512 + (lbase + (p & 15)) * 16 + i] = (__bf16)vv[u];
    }
  }
  // ---- stage q tile [256ch x 16qp] ----
  {
    int ch = tid;
    for (int qp = 0; qp < 16; ++qp)
      Ql[ch * 16 + qp] =
          qbuf[(size_t)((b * 256 + ch) * 32 + hq0 + (qp >> 2)) * 32 + wq0 + (qp & 3)];
  }
  if (tid < 16)
    Gl[tid] = gwbuf[(b * 32 + hq0 + (tid >> 2)) * 32 + wq0 + (tid & 3)];
  __syncthreads();

  // ---- K projection: 256 out-ch x 256 px, K=256 (wave: 2 mtiles x 16 ntiles)
  {
    v16bf ak[2][8];
#pragma unroll
    for (int mm = 0; mm < 2; ++mm)
#pragma unroll
      for (int fk = 0; fk < 8; ++fk)
        ak[mm][fk] =
            *(const v16bf*)&wkp[(size_t)(((2 * wvid + mm) * 8 + fk) * 32 + lane) * 16];
    float bk0[8], bk1[8];
#pragma unroll
    for (int r = 0; r < 8; ++r) {
      bk0[r] = bk[(2 * wvid) * 16 + hi * 8 + r];
      bk1[r] = bk[(2 * wvid + 1) * 16 + hi * 8 + r];
    }
    for (int nt = 0; nt < 16; ++nt) {
      v8f acc0 = {}; v8f acc1 = {};
#pragma unroll
      for (int fk = 0; fk < 8; ++fk) {
        v16bf bfrag = *(const v16bf*)&Fl[(fk * 16 + nt) * 512 + lane * 16];
        acc0 = __builtin_amdgcn_wmma_f32_16x16x32_bf16(false, ak[0][fk], false, bfrag,
                                                       (short)0, acc0, false, false);
        acc1 = __builtin_amdgcn_wmma_f32_16x16x32_bf16(false, ak[1][fk], false, bfrag,
                                                       (short)0, acc1, false, false);
      }
      int pix = nt * 16 + nlo;
#pragma unroll
      for (int r = 0; r < 8; ++r) {
        int ch0 = (2 * wvid) * 16 + hi * 8 + r;
        int ch1 = (2 * wvid + 1) * 16 + hi * 8 + r;
        KV[ch0 * 256 + pix] = (__bf16)(acc0[r] + bk0[r]);
        KV[ch1 * 256 + pix] = (__bf16)(acc1[r] + bk1[r]);
      }
    }
  }
  __syncthreads();

  // ---- attention scores + softmax (per query-pixel, per head) ----
  if (tid < 128) {
    int qp = tid >> 3, hd = tid & 7;
    int ly = qp >> 2, lx = qp & 3;
    float s[16];
    float mx = -1e30f;
#pragma unroll
    for (int pos = 0; pos < 16; ++pos) {
      int p = (ly * 4 + (pos >> 2)) * 16 + lx * 4 + (pos & 3);
      float d = 0.f;
      for (int dd = 0; dd < 32; ++dd) {
        int ch = hd * 32 + dd;
        d += Ql[ch * 16 + qp] * (float)KV[ch * 256 + p];
      }
      s[pos] = d;
      mx = fmaxf(mx, d);
    }
    float sum = 0.f;
#pragma unroll
    for (int pos = 0; pos < 16; ++pos) { s[pos] = __expf(s[pos] - mx); sum += s[pos]; }
    float inv = 1.0f / sum;
#pragma unroll
    for (int pos = 0; pos < 16; ++pos) At[(qp * 8 + hd) * 16 + pos] = s[pos] * inv;
  }
  __syncthreads();

  // ---- V projection, overwriting the K buffer ----
  {
    v16bf av[2][8];
#pragma unroll
    for (int mm = 0; mm < 2; ++mm)
#pragma unroll
      for (int fk = 0; fk < 8; ++fk)
        av[mm][fk] =
            *(const v16bf*)&wvp[(size_t)(((2 * wvid + mm) * 8 + fk) * 32 + lane) * 16];
    float bv0[8], bv1[8];
#pragma unroll
    for (int r = 0; r < 8; ++r) {
      bv0[r] = bv[(2 * wvid) * 16 + hi * 8 + r];
      bv1[r] = bv[(2 * wvid + 1) * 16 + hi * 8 + r];
    }
    for (int nt = 0; nt < 16; ++nt) {
      v8f acc0 = {}; v8f acc1 = {};
#pragma unroll
      for (int fk = 0; fk < 8; ++fk) {
        v16bf bfrag = *(const v16bf*)&Fl[(fk * 16 + nt) * 512 + lane * 16];
        acc0 = __builtin_amdgcn_wmma_f32_16x16x32_bf16(false, av[0][fk], false, bfrag,
                                                       (short)0, acc0, false, false);
        acc1 = __builtin_amdgcn_wmma_f32_16x16x32_bf16(false, av[1][fk], false, bfrag,
                                                       (short)0, acc1, false, false);
      }
      int pix = nt * 16 + nlo;
#pragma unroll
      for (int r = 0; r < 8; ++r) {
        int ch0 = (2 * wvid) * 16 + hi * 8 + r;
        int ch1 = (2 * wvid + 1) * 16 + hi * 8 + r;
        KV[ch0 * 256 + pix] = (__bf16)(acc0[r] + bv0[r]);
        KV[ch1 * 256 + pix] = (__bf16)(acc1[r] + bv1[r]);
      }
    }
  }
  __syncthreads();

  // ---- attention-weighted sum of V, gate, store (one thread per channel) ----
  {
    int ch = tid;
    int hd = ch >> 5;
    for (int qp = 0; qp < 16; ++qp) {
      int ly = qp >> 2, lx = qp & 3;
      float acc = 0.f;
#pragma unroll
      for (int pos = 0; pos < 16; ++pos) {
        int p = (ly * 4 + (pos >> 2)) * 16 + lx * 4 + (pos & 3);
        acc += At[(qp * 8 + hd) * 16 + pos] * (float)KV[ch * 256 + p];
      }
      out[(size_t)((b * 256 + ch) * 32 + hq0 + ly) * 32 + wq0 + lx] = acc * Gl[qp];
    }
  }
}

// ---------------------------------------------------------------------------
extern "C" void kernel_launch(void* const* d_in, const int* in_sizes, int n_in,
                              void* d_out, int out_size, void* d_ws, size_t ws_size,
                              hipStream_t stream) {
  const float* x  = (const float*)d_in[0];
  const float* fm = (const float*)d_in[1];
  const float* wq = (const float*)d_in[2];
  const float* bq = (const float*)d_in[3];
  const float* wk = (const float*)d_in[4];
  const float* bk = (const float*)d_in[5];
  const float* wv = (const float*)d_in[6];
  const float* bv = (const float*)d_in[7];
  const float* wg = (const float*)d_in[8];
  const float* bg = (const float*)d_in[9];

  char* ws = (char*)d_ws;
  __bf16* wkp = (__bf16*)(ws + WKP_OFF);
  __bf16* wvp = (__bf16*)(ws + WVP_OFF);
  __bf16* wqp = (__bf16*)(ws + WQP_OFF);
  float*  qb  = (float*)(ws + Q_OFF);
  float*  gwb = (float*)(ws + GW_OFF);

  pack_weights_kernel<<<768, 256, 0, stream>>>(wq, wk, wv, wkp, wvp, wqp);
  proj_q_gw_kernel<<<64, 256, 65536, stream>>>(x, wqp, bq, wg, bg, qb, gwb);
  fused_attn_kernel<<<512, 256, 286784, stream>>>(fm, wkp, wvp, bk, bv, qb, gwb,
                                                  (float*)d_out);
}
